// Harmonic2FormsNetwork_TCS_12137577578641
// MI455X (gfx1250) — matrix operand order repaired and checked by reading, not compiled, not run on hardware
//
#include <hip/hip_runtime.h>

// Harmonic 21-form MLP, fused: rotate -> fourier encode (+noise) -> 3-layer MLP
// GEMMs on v_wmma_f32_16x16x32_bf16 (bf16 operands, f32 accumulate).
// Compute-bound => everything resident in LDS; weights staged once per WG and
// reused for SUBTILES batch tiles; encoding is element-parallel across all waves.

#define N_FORMS   21
#define OUT_DIM   21
#define HID       128
#define DPOS      7
#define N_FREQ    5
#define ENC       70
#define ENC_PAD   96      // pad K to multiple of 32
#define BATCH     65536
#define ROWS_PER_WG 128   // 8 waves x 16 rows per pass
#define SUBTILES  4       // passes per workgroup (weight staging amortization)
#define BLOCK     256
#define OUT_PAD   32      // pad N of layer 3 to 2 WMMA tiles

typedef __attribute__((ext_vector_type(16))) __bf16 v16bf;
typedef __attribute__((ext_vector_type(8)))  float  v8f;

union FragU { v16bf v; uint4 q[2]; };

__device__ __forceinline__ unsigned short f2bf(float f) {
  unsigned u = __float_as_uint(f);
  u += 0x7FFFu + ((u >> 16) & 1u);          // round-to-nearest-even
  return (unsigned short)(u >> 16);
}

__device__ __forceinline__ unsigned pcg(unsigned v) {
  v = v * 747796405u + 2891336453u;
  unsigned w = ((v >> ((v >> 28) + 4u)) ^ v) * 277803737u;
  return (w >> 22) ^ w;
}

__device__ __forceinline__ float silu(float x) {
  return x * __builtin_amdgcn_rcpf(1.0f + __expf(-x));   // v_exp + v_rcp
}

// A fragment 16x32 bf16: lane(lo<16) K=k0..k0+7 & k0+16..k0+23; hi half: +8 / +24.
__device__ __forceinline__ v16bf load_a_frag(const unsigned short* rowbase, int strideB,
                                             int k0, unsigned lo, unsigned hi) {
  const char* p = (const char*)rowbase + (int)lo * strideB + k0 * 2 + (hi << 4);
  FragU f;
  f.q[0] = *(const uint4*)(p);
  f.q[1] = *(const uint4*)(p + 32);
  return f.v;
}

// B fragment 32x16 bf16 from LDS holding W transposed (Wt[n][k]):
// lane lo = column n; K=k0..k0+7 & k0+16..k0+23 (lo half), +8 / +24 (hi half).
__device__ __forceinline__ v16bf load_b_frag(const unsigned short* colbase, int strideB,
                                             int k0, unsigned lo, unsigned hi) {
  const char* p = (const char*)colbase + (int)lo * strideB + k0 * 2 + (hi << 4);
  FragU f;
  f.q[0] = *(const uint4*)(p);
  f.q[1] = *(const uint4*)(p + 32);
  return f.v;
}

__global__ __launch_bounds__(BLOCK) void harmonic_forms_wmma(
    const float* __restrict__ coords, const float* __restrict__ R,
    const float* __restrict__ W1, const float* __restrict__ b1,
    const float* __restrict__ W2, const float* __restrict__ b2,
    const float* __restrict__ W3, const float* __restrict__ b3,
    float* __restrict__ out) {
  extern __shared__ char smem[];
  unsigned short* sW1 = (unsigned short*)(smem);            // [128][96]  W1^T bf16
  unsigned short* sW2 = (unsigned short*)(smem + 24576);    // [128][128] W2^T bf16
  unsigned short* sW3 = (unsigned short*)(smem + 57344);    // [32][128]  W3^T bf16 (padded)
  unsigned short* sA  = (unsigned short*)(smem + 65536);    // [128][96]  encoded input bf16
  unsigned short* sH  = (unsigned short*)(smem + 90112);    // [128][128] activations bf16
  float* sB1 = (float*)(smem + 122880);                     // [128]
  float* sB2 = (float*)(smem + 123392);                     // [128]
  float* sB3 = (float*)(smem + 123904);                     // [32]
  float* sR  = (float*)(smem + 124032);                     // [49]
  float* sX  = (float*)(smem + 124288);                     // [128][8] rotated coords

  const int f    = blockIdx.y;
  const int tile = blockIdx.x;
  const int t    = threadIdx.x;

  // ---- stage this form's weights into LDS, transposed + bf16 (once per WG) ----
  const size_t w1off = (size_t)f * ENC * HID;
  const size_t w2off = (size_t)f * HID * HID;
  const size_t w3off = (size_t)f * HID * OUT_DIM;

  for (int idx = t; idx < HID * ENC_PAD; idx += BLOCK) {
    int n = idx / ENC_PAD, k = idx - n * ENC_PAD;
    float w = (k < ENC) ? W1[w1off + (size_t)k * HID + n] : 0.0f;
    sW1[idx] = f2bf(w);
  }
  for (int idx = t; idx < HID * HID; idx += BLOCK) {
    int n = idx >> 7, k = idx & 127;
    sW2[idx] = f2bf(W2[w2off + (size_t)k * HID + n]);
  }
  for (int idx = t; idx < OUT_PAD * HID; idx += BLOCK) {
    int n = idx >> 7, k = idx & 127;
    float w = (n < OUT_DIM) ? W3[w3off + (size_t)k * OUT_DIM + n] : 0.0f;
    sW3[idx] = f2bf(w);
  }
  if (t < HID) { sB1[t] = b1[f * HID + t]; sB2[t] = b2[f * HID + t]; }
  if (t < OUT_PAD) sB3[t] = (t < OUT_DIM) ? b3[f * OUT_DIM + t] : 0.0f;
  if (t < DPOS * DPOS) sR[t] = R[t];
  // zero sA once (padding columns stay zero across all passes)
  {
    uint4 z = {0u, 0u, 0u, 0u};
    uint4* p = (uint4*)sA;
    for (int idx = t; idx < (ROWS_PER_WG * ENC_PAD * 2) / 16; idx += BLOCK) p[idx] = z;
  }
  __syncthreads();

  const unsigned lane = (unsigned)t & 31u;
  const unsigned wave = (unsigned)t >> 5;
  const unsigned lo   = lane & 15u;
  const unsigned hi   = lane >> 4;
  const v8f vzero = {0.f, 0.f, 0.f, 0.f, 0.f, 0.f, 0.f, 0.f};
  const float nscale = 0.01f * (float)(f + 1) / (float)N_FORMS;
  const unsigned fseed = (unsigned)(f * 0x01000193u);

  const unsigned short* aBase = sA + wave * 16 * ENC_PAD;
  unsigned short* hBase = sH + wave * 16 * HID;

#pragma unroll 1
  for (int st = 0; st < SUBTILES; ++st) {
    const int rowBase = (tile * SUBTILES + st) * ROWS_PER_WG;

    // ---- encode phase A: x = c @ R^T per row (cheap, 49 FMA) ----
    if (t < ROWS_PER_WG) {
      float c[DPOS];
#pragma unroll
      for (int j = 0; j < DPOS; ++j) c[j] = coords[(size_t)(rowBase + t) * DPOS + j];
#pragma unroll
      for (int d = 0; d < DPOS; ++d) {
        float x = 0.0f;
#pragma unroll
        for (int j = 0; j < DPOS; ++j) x += c[j] * sR[d * DPOS + j];
        sX[t * 8 + d] = x;
      }
    }
    __syncthreads();

    // ---- encode phase B: element-parallel sin/cos + noise over all 256 threads ----
    // feat[e], e = d*10 + k (sin) / d*10+5+k (cos); cos(a) = sin(a + pi/2).
    for (int idx = t; idx < ROWS_PER_WG * ENC; idx += BLOCK) {
      int r = idx / ENC;
      int e = idx - r * ENC;
      int d = e / 10;
      int m = e - d * 10;
      bool isCos = (m >= 5);
      int k = isCos ? m - 5 : m;
      float ang = ldexpf(sX[r * 8 + d], k) + (isCos ? 1.5707963267948966f : 0.0f);
      float v = __sinf(ang);
      unsigned h = pcg(((unsigned)(rowBase + r) * 0x9E3779B9u) ^
                       ((unsigned)e * 0x85EBCA77u) ^ fseed);
      // triangular approx of N(0,1): (u1-u2)*sqrt(6), u 16-bit uniforms
      float noise = ((float)(h & 0xFFFFu) - (float)(h >> 16)) *
                    (2.4494897f / 65536.0f);
      sA[r * ENC_PAD + e] = f2bf(v + noise * nscale);
    }
    __syncthreads();

    // ---- per-wave WMMA pipeline: each wave owns 16 batch rows ----
    // Layer 1: (16 x 96) @ (96 x 128) + b1, SiLU -> sH
    v8f acc[8];
#pragma unroll
    for (int j = 0; j < 8; ++j) acc[j] = vzero;
#pragma unroll
    for (int ks = 0; ks < 3; ++ks) {
      v16bf a = load_a_frag(aBase, ENC_PAD * 2, ks * 32, lo, hi);
#pragma unroll
      for (int j = 0; j < 8; ++j) {
        v16bf b = load_b_frag(sW1 + j * 16 * ENC_PAD, ENC_PAD * 2, ks * 32, lo, hi);
        acc[j] = __builtin_amdgcn_wmma_f32_16x16x32_bf16(false, a, false, b,
                                                         (short)0, acc[j], false, false);
      }
    }
#pragma unroll
    for (int j = 0; j < 8; ++j) {
      float bias = sB1[16 * j + (int)lo];
#pragma unroll
      for (int v = 0; v < 8; ++v) {
        // C layout: row = v + 8*hi, col = 16j+lo
        hBase[(v + 8 * (int)hi) * HID + 16 * j + (int)lo] = f2bf(silu(acc[j][v] + bias));
      }
    }
    // Wave reads/writes only its own sH region; same-wave LDS ops are in-order.

    // Layer 2: (16 x 128) @ (128 x 128) + b2, SiLU -> sH (in place)
#pragma unroll
    for (int j = 0; j < 8; ++j) acc[j] = vzero;
#pragma unroll
    for (int ks = 0; ks < 4; ++ks) {
      v16bf a = load_a_frag(hBase, HID * 2, ks * 32, lo, hi);
#pragma unroll
      for (int j = 0; j < 8; ++j) {
        v16bf b = load_b_frag(sW2 + j * 16 * HID, HID * 2, ks * 32, lo, hi);
        acc[j] = __builtin_amdgcn_wmma_f32_16x16x32_bf16(false, a, false, b,
                                                         (short)0, acc[j], false, false);
      }
    }
#pragma unroll
    for (int j = 0; j < 8; ++j) {
      float bias = sB2[16 * j + (int)lo];
#pragma unroll
      for (int v = 0; v < 8; ++v) {
        hBase[(v + 8 * (int)hi) * HID + 16 * j + (int)lo] = f2bf(silu(acc[j][v] + bias));
      }
    }

    // Layer 3: (16 x 128) @ (128 x 32) + b3 -> global (B, 21, 21) f32
    v8f acc3[2];
    acc3[0] = vzero; acc3[1] = vzero;
#pragma unroll
    for (int ks = 0; ks < 4; ++ks) {
      v16bf a = load_a_frag(hBase, HID * 2, ks * 32, lo, hi);
#pragma unroll
      for (int j = 0; j < 2; ++j) {
        v16bf b = load_b_frag(sW3 + j * 16 * HID, HID * 2, ks * 32, lo, hi);
        acc3[j] = __builtin_amdgcn_wmma_f32_16x16x32_bf16(false, a, false, b,
                                                          (short)0, acc3[j], false, false);
      }
    }
    const int wrow = rowBase + (int)wave * 16;
#pragma unroll
    for (int j = 0; j < 2; ++j) {
      int o = 16 * j + (int)lo;
      if (o < OUT_DIM) {
        float bias = sB3[o];
#pragma unroll
        for (int v = 0; v < 8; ++v) {
          int gb = wrow + v + 8 * (int)hi;
          out[(size_t)gb * (N_FORMS * OUT_DIM) + f * OUT_DIM + o] = acc3[j][v] + bias;
        }
      }
    }
    __syncthreads();   // pass p fully consumed sA/sX before pass p+1 rewrites them
  }
}

extern "C" void kernel_launch(void* const* d_in, const int* in_sizes, int n_in,
                              void* d_out, int out_size, void* d_ws, size_t ws_size,
                              hipStream_t stream) {
  (void)in_sizes; (void)n_in; (void)out_size; (void)d_ws; (void)ws_size;
  const float* coords = (const float*)d_in[0];
  const float* R  = (const float*)d_in[1];
  const float* W1 = (const float*)d_in[2];
  const float* b1 = (const float*)d_in[3];
  const float* W2 = (const float*)d_in[4];
  const float* b2 = (const float*)d_in[5];
  const float* W3 = (const float*)d_in[6];
  const float* b3 = (const float*)d_in[7];
  float* out = (float*)d_out;

  dim3 grid(BATCH / (ROWS_PER_WG * SUBTILES), N_FORMS, 1);  // 128 x 21 workgroups
  dim3 block(BLOCK, 1, 1);                                  // 8 wave32s
  const size_t smem = 128384;                               // dynamic LDS carve-out
  hipLaunchKernelGGL(harmonic_forms_wmma, grid, block, smem, stream,
                     coords, R, W1, b1, W2, b2, W3, b3, out);
}
